// GLAAttention_31379031065181
// MI455X (gfx1250) — matrix operand order repaired and checked by reading, compiled, not run on hardware
//
#include <hip/hip_runtime.h>
#include <math.h>

typedef __attribute__((ext_vector_type(16))) __bf16 v16bf;
typedef __attribute__((ext_vector_type(8)))  float  v8f;
typedef __attribute__((ext_vector_type(4)))  int    vi4;

#if defined(__HIP_DEVICE_COMPILE__) && __has_builtin(__builtin_amdgcn_global_load_async_to_lds_b128)
#define HAVE_ASYNC 1
#else
#define HAVE_ASYNC 0
#endif

__device__ __forceinline__ void async_wait0() {
#if HAVE_ASYNC
#if __has_builtin(__builtin_amdgcn_s_wait_asynccnt)
  __builtin_amdgcn_s_wait_asynccnt(0);
#else
  asm volatile("s_wait_asynccnt 0x0" ::: "memory");
#endif
#endif
}

// 16-byte global -> LDS copy (async DMA path on CDNA5, sync fallback)
__device__ __forceinline__ void copy16_g2l(const unsigned short* g, unsigned short* l) {
#if HAVE_ASYNC
  __builtin_amdgcn_global_load_async_to_lds_b128((vi4*)g, (vi4*)l, 0, 0);
#else
  *(uint4*)l = *(const uint4*)g;
#endif
}

__device__ __forceinline__ unsigned short f2bf(float f) {
  unsigned int u = __float_as_uint(f);
  u += 0x7FFFu + ((u >> 16) & 1u);          // round-to-nearest-even
  return (unsigned short)(u >> 16);
}

union ABFrag { v16bf v; uint4 q[2]; };

#define BM 128
#define BN 128
#define BK 32
#define LDP 40              // padded halfword row stride (80B, 16B-aligned)
#define TILE_US (128 * LDP)

// Stage one 128x32 bf16 tile (rows_valid rows) into LDS, zero-filling OOB.
__device__ __forceinline__ void stage_tile(const unsigned short* __restrict__ src,
                                           unsigned short* lds, int rows_valid,
                                           int K, int k0, int tid) {
  const uint4 z = {0u, 0u, 0u, 0u};
#pragma unroll
  for (int h = 0; h < 2; ++h) {
    int c = tid + h * 256;          // 512 16B chunks per tile
    int row = c >> 2, part = c & 3;
    int kb = k0 + part * 8;
    unsigned short* dst = lds + row * LDP + part * 8;
    if (row < rows_valid && kb + 8 <= K)
      copy16_g2l(src + (size_t)row * K + kb, dst);
    else
      *(uint4*)dst = z;
  }
}

// C[M,N] = act( A[M,K](bf16) @ Bt[N,K](bf16)^T ), fp32 accumulate.
// act: 0 = identity, 1 = *scale, 2 = logsigmoid(v + bias[n]) / 16
// out_bf16: 1 -> C is bf16 (ushort), else fp32.
__global__ __launch_bounds__(256) void wmma_gemm_bf16(
    const unsigned short* __restrict__ A, const unsigned short* __restrict__ Bt,
    void* __restrict__ Cout, int M, int N, int K,
    float scale, const float* __restrict__ bias, int act, int out_bf16)
{
  __shared__ __align__(16) unsigned short ldsA[2 * TILE_US];
  __shared__ __align__(16) unsigned short ldsB[2 * TILE_US];
  const int tid  = threadIdx.x;
  const int m0   = blockIdx.y * BM;
  const int n0   = blockIdx.x * BN;
  const int lane = tid & 31;
  const int w    = tid >> 5;
  const int wm   = w & 3;    // 4 waves along M: 32 rows each
  const int wn   = w >> 2;   // 2 waves along N: 64 cols each

  v8f acc[2][4];
  const v8f vzero = {0.f,0.f,0.f,0.f,0.f,0.f,0.f,0.f};
#pragma unroll
  for (int i = 0; i < 2; ++i)
#pragma unroll
    for (int j = 0; j < 4; ++j) acc[i][j] = vzero;

  const int browsv = (N - n0 < BN) ? (N - n0) : BN;

  // prologue stage into buffer 0
  stage_tile(A  + (size_t)m0 * K, ldsA, BM,     K, 0, tid);
  stage_tile(Bt + (size_t)n0 * K, ldsB, browsv, K, 0, tid);

  const int frow = lane & 15;
  const int kbA  = (lane >> 4) * 8;    // A frag: K chunks {kbA..+7},{kbA+16..+23}
  const int kbB  = (lane >> 4) * 8;    // B frag: K chunks {kbB..+7},{kbB+16..+23}
  int buf = 0;

  for (int k0 = 0; k0 < K; k0 += BK) {
    async_wait0();
    __syncthreads();
    // prefetch next tile into the other buffer while computing this one
    if (k0 + BK < K) {
      stage_tile(A  + (size_t)m0 * K, ldsA + (buf ^ 1) * TILE_US, BM,     K, k0 + BK, tid);
      stage_tile(Bt + (size_t)n0 * K, ldsB + (buf ^ 1) * TILE_US, browsv, K, k0 + BK, tid);
    }
    const unsigned short* a_ = ldsA + buf * TILE_US;
    const unsigned short* b_ = ldsB + buf * TILE_US;

    ABFrag af[2];
#pragma unroll
    for (int i = 0; i < 2; ++i) {
      int r = wm * 32 + i * 16 + frow;
      af[i].q[0] = *(const uint4*)&a_[r * LDP + kbA];
      af[i].q[1] = *(const uint4*)&a_[r * LDP + kbA + 16];
    }
#pragma unroll
    for (int j = 0; j < 4; ++j) {
      ABFrag bf_;
      int c = wn * 64 + j * 16 + frow;
      bf_.q[0] = *(const uint4*)&b_[c * LDP + kbB];
      bf_.q[1] = *(const uint4*)&b_[c * LDP + kbB + 16];
#pragma unroll
      for (int i = 0; i < 2; ++i)
        acc[i][j] = __builtin_amdgcn_wmma_f32_16x16x32_bf16(
            false, af[i].v, false, bf_.v, (short)0, acc[i][j], false, false);
    }
    buf ^= 1;
  }

  // epilogue: C layout lane<16 -> M=r,N=lane ; lane>=16 -> M=r+8,N=lane-16
  const int nlane = lane & 15;
  const int mhigh = (lane >> 4) * 8;
#pragma unroll
  for (int i = 0; i < 2; ++i) {
#pragma unroll
    for (int j = 0; j < 4; ++j) {
      int n = n0 + wn * 64 + j * 16 + nlane;
      if (n >= N) continue;
#pragma unroll
      for (int r = 0; r < 8; ++r) {
        int m = m0 + wm * 32 + i * 16 + mhigh + r;
        float vv = acc[i][j][r];
        if (act == 1) {
          vv *= scale;
        } else if (act == 2) {
          float z  = vv + bias[n];
          float ls = fminf(z, 0.f) - log1pf(__expf(-fabsf(z)));
          vv = ls * (1.0f / 16.0f);
        }
        if (out_bf16) ((unsigned short*)Cout)[(size_t)m * N + n] = f2bf(vv);
        else          ((float*)Cout)[(size_t)m * N + n] = vv;
      }
    }
  }
}

// elementwise fp32 -> bf16
__global__ __launch_bounds__(256) void cvt_f32_bf16(const float* __restrict__ in,
                                                    unsigned short* __restrict__ out,
                                                    int n) {
  int i = (blockIdx.x * 256 + threadIdx.x) * 4;
  if (i + 4 <= n) {
    float4 f = *(const float4*)(in + i);
    ushort4 u;
    u.x = f2bf(f.x); u.y = f2bf(f.y); u.z = f2bf(f.z); u.w = f2bf(f.w);
    *(ushort4*)(out + i) = u;
  } else {
    for (; i < n; ++i) out[i] = f2bf(in[i]);
  }
}

// W[K,N] fp32 -> Wt[N,K] bf16 (32x32 LDS tile transpose)
__global__ __launch_bounds__(256) void transpose_cvt_bf16(
    const float* __restrict__ W, unsigned short* __restrict__ Wt, int K, int N) {
  __shared__ unsigned short tile[32][33];
  const int k0 = blockIdx.y * 32, n0 = blockIdx.x * 32;
  const int tx = threadIdx.x & 31, ty = threadIdx.x >> 5;  // ty: 0..7
#pragma unroll
  for (int i = 0; i < 32; i += 8) {
    int kk = k0 + ty + i, n = n0 + tx;
    float f = (kk < K && n < N) ? W[(size_t)kk * N + n] : 0.f;
    tile[ty + i][tx] = f2bf(f);
  }
  __syncthreads();
#pragma unroll
  for (int i = 0; i < 32; i += 8) {
    int n = n0 + ty + i, kk = k0 + tx;
    if (n < N && kk < K) Wt[(size_t)n * K + kk] = tile[tx][ty + i];
  }
}

// Sequential GLA scan; S columns independent -> grid (DV/16, H, B), wave32 blocks.
__global__ __launch_bounds__(32) void gla_scan(
    const float* __restrict__ q, const float* __restrict__ k,
    const float* __restrict__ v, const float* __restrict__ gk,
    float* __restrict__ o)
{
  const int vc = blockIdx.x, h = blockIdx.y, b = blockIdx.z;
  const int lane  = threadIdx.x;
  const int vlane = lane & 15;
  const int khalf = lane >> 4;
  const int kbase = khalf * 32;
  __shared__ float qs[64], ks[64], egs[64], vs[16];
  float s[32];
#pragma unroll
  for (int i = 0; i < 32; ++i) s[i] = 0.f;

  const float* qp = q  + (size_t)b * 2048 * 1024 + h * 64;
  const float* kp = k  + (size_t)b * 2048 * 1024 + h * 64;
  const float* gp = gk + (size_t)b * 2048 * 1024 + h * 64;
  const float* vp = v  + (size_t)b * 2048 * 2048 + h * 128 + vc * 16;
  float*       op = o  + (size_t)b * 2048 * 2048 + h * 128 + vc * 16;

  for (int t = 0; t < 2048; ++t) {
    qs[lane]       = qp[lane];
    qs[lane + 32]  = qp[lane + 32];
    ks[lane]       = kp[lane];
    ks[lane + 32]  = kp[lane + 32];
    egs[lane]      = __expf(gp[lane]);
    egs[lane + 32] = __expf(gp[lane + 32]);
    if (lane < 16) vs[lane] = vp[lane];
    __syncthreads();

    float vv = vs[vlane];
    float part = 0.f;
#pragma unroll
    for (int kk = 0; kk < 32; ++kk) {
      float sn = fmaf(ks[kbase + kk], vv, s[kk] * egs[kbase + kk]);
      s[kk] = sn;
      part = fmaf(qs[kbase + kk], sn, part);
    }
    part += __shfl_xor(part, 16, 32);
    if (khalf == 0) op[vlane] = part;
    __syncthreads();

    qp += 1024; kp += 1024; gp += 1024; vp += 2048; op += 2048;
  }
}

// Fused per-head RMSNorm * norm_w * swish(g); writes bf16 for the Wo GEMM.
__global__ __launch_bounds__(256) void gla_gate_norm(
    const float* __restrict__ o, const float* __restrict__ g,
    const float* __restrict__ norm_w, unsigned short* __restrict__ ybf)
{
  const int lane = threadIdx.x & 31;
  const size_t row = (size_t)blockIdx.x * 8 + (threadIdx.x >> 5);
  const float* orow = o + row * 128;
  const float* grow = g + row * 128;

  float4 ov = *(const float4*)(orow + lane * 4);
  float ss = ov.x*ov.x + ov.y*ov.y + ov.z*ov.z + ov.w*ov.w;
#pragma unroll
  for (int off = 16; off; off >>= 1) ss += __shfl_xor(ss, off, 32);
  float r = rsqrtf(ss * (1.0f / 128.0f) + 1e-5f);

  float4 gv = *(const float4*)(grow + lane * 4);
  float4 wv = *(const float4*)(norm_w + lane * 4);
  ushort4 y;
  y.x = f2bf(ov.x * r * wv.x * (gv.x / (1.f + __expf(-gv.x))));
  y.y = f2bf(ov.y * r * wv.y * (gv.y / (1.f + __expf(-gv.y))));
  y.z = f2bf(ov.z * r * wv.z * (gv.z / (1.f + __expf(-gv.z))));
  y.w = f2bf(ov.w * r * wv.w * (gv.w / (1.f + __expf(-gv.w))));
  *(ushort4*)(ybf + row * 128 + lane * 4) = y;
}

extern "C" void kernel_launch(void* const* d_in, const int* in_sizes, int n_in,
                              void* d_out, int out_size, void* d_ws, size_t ws_size,
                              hipStream_t stream) {
  (void)in_sizes; (void)n_in; (void)out_size; (void)ws_size;
  const float* x      = (const float*)d_in[0];
  const float* Wq     = (const float*)d_in[1];
  const float* Wk     = (const float*)d_in[2];
  const float* Wv     = (const float*)d_in[3];
  const float* Wg     = (const float*)d_in[4];
  const float* Wgk1   = (const float*)d_in[5];
  const float* Wgk2   = (const float*)d_in[6];
  const float* bgk2   = (const float*)d_in[7];
  const float* norm_w = (const float*)d_in[8];
  const float* Wo     = (const float*)d_in[9];

  // fp32 intermediates
  float* fws = (float*)d_ws;
  float* q   = fws;                   // 8192*1024
  float* kk  = q  + 8388608;          // 8192*1024
  float* gk  = kk + 8388608;          // 8192*1024
  float* v   = gk + 8388608;          // 8192*2048
  float* g   = v  + 16777216;         // 8192*2048
  float* o   = g  + 16777216;         // 8192*2048
  // bf16 region
  unsigned short* us    = (unsigned short*)(fws + 75497472);
  unsigned short* xbf   = us;                    // 8192*2048
  unsigned short* WqT   = xbf   + 16777216;      // 1024*2048
  unsigned short* WkT   = WqT   + 2097152;
  unsigned short* WvT   = WkT   + 2097152;       // 2048*2048
  unsigned short* WgT   = WvT   + 4194304;
  unsigned short* WoT   = WgT   + 4194304;
  unsigned short* Wgk1T = WoT   + 4194304;       // 16*2048
  unsigned short* Wgk2T = Wgk1T + 32768;         // 1024*16
  unsigned short* tmpbf = Wgk2T + 16384;         // 8192*16
  unsigned short* obf   = tmpbf + 131072;        // 8192*2048

  dim3 blk(256);
  // one-time bf16 conversion / weight transposition
  cvt_f32_bf16<<<dim3(16384), blk, 0, stream>>>(x, xbf, 16777216);
  transpose_cvt_bf16<<<dim3(32, 64), blk, 0, stream>>>(Wq,   WqT,   2048, 1024);
  transpose_cvt_bf16<<<dim3(32, 64), blk, 0, stream>>>(Wk,   WkT,   2048, 1024);
  transpose_cvt_bf16<<<dim3(64, 64), blk, 0, stream>>>(Wv,   WvT,   2048, 2048);
  transpose_cvt_bf16<<<dim3(64, 64), blk, 0, stream>>>(Wg,   WgT,   2048, 2048);
  transpose_cvt_bf16<<<dim3(64, 64), blk, 0, stream>>>(Wo,   WoT,   2048, 2048);
  transpose_cvt_bf16<<<dim3(1, 64),  blk, 0, stream>>>(Wgk1, Wgk1T, 2048, 16);
  transpose_cvt_bf16<<<dim3(32, 1),  blk, 0, stream>>>(Wgk2, Wgk2T, 16,   1024);

  // projections (bf16 WMMA, fp32 accumulate)
  wmma_gemm_bf16<<<dim3(8, 64),  blk, 0, stream>>>(xbf, WqT, q,  8192, 1024, 2048, 0.125f, nullptr, 1, 0);
  wmma_gemm_bf16<<<dim3(8, 64),  blk, 0, stream>>>(xbf, WkT, kk, 8192, 1024, 2048, 1.f,    nullptr, 0, 0);
  wmma_gemm_bf16<<<dim3(16, 64), blk, 0, stream>>>(xbf, WvT, v,  8192, 2048, 2048, 1.f,    nullptr, 0, 0);
  wmma_gemm_bf16<<<dim3(16, 64), blk, 0, stream>>>(xbf, WgT, g,  8192, 2048, 2048, 1.f,    nullptr, 0, 0);
  // low-rank forget gate: tmp = x@Wgk1 (bf16 out) ; gk = logsigmoid(tmp@Wgk2 + b)/16
  wmma_gemm_bf16<<<dim3(1, 64),  blk, 0, stream>>>(xbf,   Wgk1T, tmpbf, 8192, 16,   2048, 1.f, nullptr, 0, 1);
  wmma_gemm_bf16<<<dim3(8, 64),  blk, 0, stream>>>(tmpbf, Wgk2T, gk,    8192, 1024, 16,   1.f, bgk2,    2, 0);
  // recurrent scan, fused gate+norm (-> bf16), output projection
  gla_scan<<<dim3(8, 16, 4), dim3(32), 0, stream>>>(q, kk, v, gk, o);
  gla_gate_norm<<<dim3(16384), blk, 0, stream>>>(o, g, norm_w, obf);
  wmma_gemm_bf16<<<dim3(16, 64), blk, 0, stream>>>(obf, WoT, (float*)d_out, 8192, 2048, 2048, 1.f, nullptr, 0, 0);
}
